// SpatialSelfAttention_4758823764180
// MI455X (gfx1250) — matrix operand order repaired and checked by reading, not compiled, not run on hardware
//
#include <hip/hip_runtime.h>
#include <cmath>

typedef __attribute__((ext_vector_type(16))) _Float16 v16h;
typedef __attribute__((ext_vector_type(8)))  _Float16 v8h;
typedef __attribute__((ext_vector_type(8)))  float    v8f;

// ---------------------------------------------------------------------------
// Elementwise f32 -> f16 conversion (weights)
// ---------------------------------------------------------------------------
__global__ __launch_bounds__(256) void f32_to_f16_kernel(const float* __restrict__ src,
                                                         _Float16* __restrict__ dst, int n) {
    int i = blockIdx.x * 1024 + threadIdx.x * 4;
#pragma unroll
    for (int t = 0; t < 4; ++t) {
        int j = i + t;
        if (j < n) dst[j] = (_Float16)src[j];
    }
}

// ---------------------------------------------------------------------------
// GroupNorm statistics: one block per (batch, group); group = 16 contiguous
// channels * 4096 pixels = 65536 contiguous floats.
// ---------------------------------------------------------------------------
__global__ __launch_bounds__(256) void gn_stats_kernel(const float* __restrict__ x,
                                                       float* __restrict__ mean,
                                                       float* __restrict__ rstd) {
    const int bg = blockIdx.x;
    const float* p = x + (size_t)bg * 65536u;
    float s = 0.f, s2 = 0.f;
    for (int i = threadIdx.x; i < 65536; i += 256) {
        float v = p[i];
        s += v;
        s2 += v * v;
    }
    __shared__ float rs[256], rq[256];
    rs[threadIdx.x] = s;
    rq[threadIdx.x] = s2;
    __syncthreads();
    for (int k = 128; k > 0; k >>= 1) {
        if (threadIdx.x < k) {
            rs[threadIdx.x] += rs[threadIdx.x + k];
            rq[threadIdx.x] += rq[threadIdx.x + k];
        }
        __syncthreads();
    }
    if (threadIdx.x == 0) {
        float mu = rs[0] * (1.0f / 65536.0f);
        float var = rq[0] * (1.0f / 65536.0f) - mu * mu;
        mean[bg] = mu;
        rstd[bg] = rsqrtf(var + 1e-6f);
    }
}

// ---------------------------------------------------------------------------
// GroupNorm apply + affine + transpose: x[b,c,p] (f32) -> hT[b,p,c] (f16).
// 32x32 tiles through LDS so both global sides are coalesced.
// grid = (HW/32, C/32, B), block = 256
// ---------------------------------------------------------------------------
__global__ __launch_bounds__(256) void gn_apply_transpose_kernel(
    const float* __restrict__ x, const float* __restrict__ gamma,
    const float* __restrict__ beta, const float* __restrict__ mean,
    const float* __restrict__ rstd, _Float16* __restrict__ hT) {
    constexpr int C = 512, HW = 4096;
    const int b = blockIdx.z;
    const int p0 = blockIdx.x * 32;
    const int c0 = blockIdx.y * 32;
    const int tx = threadIdx.x & 31;
    const int ty = threadIdx.x >> 5;  // 0..7
    __shared__ float tile[32][33];
    const float* xb = x + (size_t)b * C * HW;
#pragma unroll
    for (int j = 0; j < 4; ++j) {
        int c = c0 + ty + j * 8;
        int bg = b * 32 + (c >> 4);  // 16 channels per group
        float v = (xb[(size_t)c * HW + p0 + tx] - mean[bg]) * rstd[bg] * gamma[c] + beta[c];
        tile[ty + j * 8][tx] = v;  // tile[c_local][p_local]
    }
    __syncthreads();
    _Float16* hb = hT + (size_t)b * HW * C;
#pragma unroll
    for (int j = 0; j < 4; ++j) {
        int p = p0 + ty + j * 8;
        hb[(size_t)p * C + c0 + tx] = (_Float16)tile[tx][ty + j * 8];
    }
}

// ---------------------------------------------------------------------------
// Row softmax over f16 attention logits, in place. One block per row (4096).
// ---------------------------------------------------------------------------
__global__ __launch_bounds__(256) void softmax_rows_kernel(_Float16* __restrict__ attn, int ncols) {
    _Float16* p = attn + (size_t)blockIdx.x * (size_t)ncols;
    __shared__ float red[256];
    float m = -3.4e38f;
    for (int j = threadIdx.x; j < ncols; j += 256) m = fmaxf(m, (float)p[j]);
    red[threadIdx.x] = m;
    __syncthreads();
    for (int k = 128; k > 0; k >>= 1) {
        if (threadIdx.x < k) red[threadIdx.x] = fmaxf(red[threadIdx.x], red[threadIdx.x + k]);
        __syncthreads();
    }
    m = red[0];
    __syncthreads();
    float s = 0.f;
    for (int j = threadIdx.x; j < ncols; j += 256) s += __expf((float)p[j] - m);
    red[threadIdx.x] = s;
    __syncthreads();
    for (int k = 128; k > 0; k >>= 1) {
        if (threadIdx.x < k) red[threadIdx.x] += red[threadIdx.x + k];
        __syncthreads();
    }
    float inv = 1.0f / red[0];
    for (int j = threadIdx.x; j < ncols; j += 256)
        p[j] = (_Float16)(__expf((float)p[j] - m) * inv);
}

// ---------------------------------------------------------------------------
// Wave-level WMMA GEMM, B always given as Bt[n, k] row-major (ld = ldb):
//   D[m,n] = scale * sum_k A[m,k] * Bt[n,k]  (+ bias[m]) (+ resid[m*ldo+n])
// Each wave computes a 32(M) x 64(N) tile: 2 A fragments x 4 B fragments,
// 8 v_wmma per 32-wide K step. blockDim=256 (8 waves),
// grid = (N/64, M/256, batch).
// ---------------------------------------------------------------------------
template <bool OUT_F32, bool OUT_TRANS, bool ADD_BIAS, bool ADD_RES>
__global__ __launch_bounds__(256) void wmma_gemm_kernel(
    const _Float16* __restrict__ A, int lda, unsigned long long strideA,
    const _Float16* __restrict__ Bt, int ldb, unsigned long long strideB,
    void* __restrict__ Out, int ldo, unsigned long long strideO,
    const float* __restrict__ bias,
    const float* __restrict__ resid, unsigned long long strideR,
    float scale, int K) {
    const int lane = threadIdx.x & 31;
    const int wave = threadIdx.x >> 5;
    const int col = lane & 15;   // N column (B/C/D) or M row (A) within tile
    const int half = lane >> 4;  // selects K/M sub-block per ISA fragment layout
    const int m0 = (blockIdx.y * 8 + wave) * 32;
    const int n0 = blockIdx.x * 64;
    const unsigned long long z = blockIdx.z;

    const _Float16* arow0 = A + z * strideA + (size_t)(m0 + col) * (size_t)lda;
    const _Float16* arow1 = arow0 + (size_t)16 * (size_t)lda;
    const _Float16* Bb = Bt + z * strideB;

    v8f acc[2][4] = {{v8f{}, v8f{}, v8f{}, v8f{}}, {v8f{}, v8f{}, v8f{}, v8f{}}};

    for (int k0 = 0; k0 < K; k0 += 32) {
        const int kk = k0 + 8 * half;
        // A fragments: lane(M) holds K = k0 + 8*half + {0..7} and +16
        v8h a0lo = *(const v8h*)(arow0 + kk);
        v8h a0hi = *(const v8h*)(arow0 + kk + 16);
        v8h a1lo = *(const v8h*)(arow1 + kk);
        v8h a1hi = *(const v8h*)(arow1 + kk + 16);
        v16h a0 = __builtin_shufflevector(a0lo, a0hi, 0, 1, 2, 3, 4, 5, 6, 7,
                                          8, 9, 10, 11, 12, 13, 14, 15);
        v16h a1 = __builtin_shufflevector(a1lo, a1hi, 0, 1, 2, 3, 4, 5, 6, 7,
                                          8, 9, 10, 11, 12, 13, 14, 15);
        __builtin_prefetch(arow0 + kk + 32, 0, 3);
        __builtin_prefetch(arow1 + kk + 32, 0, 3);
#pragma unroll
        for (int t = 0; t < 4; ++t) {
            const int n = n0 + t * 16 + col;
            // B fragment: lane(N=n) holds K = k0 + 8*half + {0..7} and +16
            const _Float16* bp = Bb + (size_t)n * (size_t)ldb + k0 + 8 * half;
            v8h blo = *(const v8h*)bp;
            v8h bhi = *(const v8h*)(bp + 16);
            v16h b = __builtin_shufflevector(blo, bhi, 0, 1, 2, 3, 4, 5, 6, 7,
                                             8, 9, 10, 11, 12, 13, 14, 15);
            __builtin_prefetch(bp + 32, 0, 3);
            acc[0][t] = __builtin_amdgcn_wmma_f32_16x16x32_f16(
                false, a0, false, b, (short)0, acc[0][t], false, false);
            acc[1][t] = __builtin_amdgcn_wmma_f32_16x16x32_f16(
                false, a1, false, b, (short)0, acc[1][t], false, false);
        }
    }

    const float* rb = resid + z * strideR;
#pragma unroll
    for (int i = 0; i < 2; ++i) {
#pragma unroll
        for (int t = 0; t < 4; ++t) {
            const int n = n0 + t * 16 + col;
#pragma unroll
            for (int r = 0; r < 8; ++r) {
                const int m = m0 + i * 16 + r + 8 * half;  // C/D layout: VGPR r -> M=r+8*half
                float v = acc[i][t][r] * scale;
                if (ADD_BIAS) v += bias[m];
                if (ADD_RES) v += rb[(size_t)m * (size_t)ldo + n];
                const size_t oidx = OUT_TRANS ? ((size_t)n * (size_t)ldo + m)
                                              : ((size_t)m * (size_t)ldo + n);
                if (OUT_F32)
                    ((float*)Out)[z * strideO + oidx] = v;
                else
                    ((_Float16*)Out)[z * strideO + oidx] = (_Float16)v;
            }
        }
    }
}

// ---------------------------------------------------------------------------
extern "C" void kernel_launch(void* const* d_in, const int* in_sizes, int n_in,
                              void* d_out, int out_size, void* d_ws, size_t ws_size,
                              hipStream_t stream) {
    (void)in_sizes; (void)n_in; (void)out_size; (void)ws_size;
    constexpr int Bn = 2, C = 512, HW = 4096, G = 32;
    typedef unsigned long long ull;

    const float* x     = (const float*)d_in[0];
    const float* gamma = (const float*)d_in[1];
    const float* beta  = (const float*)d_in[2];
    const float* wq = (const float*)d_in[3];
    const float* bq = (const float*)d_in[4];
    const float* wk = (const float*)d_in[5];
    const float* bk = (const float*)d_in[6];
    const float* wv = (const float*)d_in[7];
    const float* bv = (const float*)d_in[8];
    const float* wo = (const float*)d_in[9];
    const float* bo = (const float*)d_in[10];

    char* ws = (char*)d_ws;
    size_t off = 0;
    auto take = [&](size_t bytes) -> char* {
        char* p = ws + off;
        off += (bytes + 255) & ~(size_t)255;
        return p;
    };
    float* meanb = (float*)take(Bn * G * sizeof(float));
    float* rstdb = (float*)take(Bn * G * sizeof(float));
    _Float16* wq16 = (_Float16*)take((size_t)C * C * 2);
    _Float16* wk16 = (_Float16*)take((size_t)C * C * 2);
    _Float16* wv16 = (_Float16*)take((size_t)C * C * 2);
    _Float16* wo16 = (_Float16*)take((size_t)C * C * 2);
    _Float16* hT   = (_Float16*)take((size_t)Bn * HW * C * 2);  // [B,HW,C]
    _Float16* qT   = (_Float16*)take((size_t)Bn * HW * C * 2);  // [B,HW,C]
    _Float16* kT   = (_Float16*)take((size_t)Bn * HW * C * 2);  // [B,HW,C]
    _Float16* vbuf = (_Float16*)take((size_t)Bn * C * HW * 2);  // [B,C,HW]
    _Float16* aoT  = (_Float16*)take((size_t)Bn * HW * C * 2);  // [B,HW,C]
    _Float16* attn = (_Float16*)take((size_t)Bn * HW * HW * 2); // [B,HW,HW]

    // 1) Weights -> f16
    f32_to_f16_kernel<<<(C * C) / 1024, 256, 0, stream>>>(wq, wq16, C * C);
    f32_to_f16_kernel<<<(C * C) / 1024, 256, 0, stream>>>(wk, wk16, C * C);
    f32_to_f16_kernel<<<(C * C) / 1024, 256, 0, stream>>>(wv, wv16, C * C);
    f32_to_f16_kernel<<<(C * C) / 1024, 256, 0, stream>>>(wo, wo16, C * C);

    // 2) GroupNorm -> hT [B, HW, C] f16
    gn_stats_kernel<<<Bn * G, 256, 0, stream>>>(x, meanb, rstdb);
    gn_apply_transpose_kernel<<<dim3(HW / 32, C / 32, Bn), 256, 0, stream>>>(
        x, gamma, beta, meanb, rstdb, hT);

    // 3) Q/K/V projections: D[o,p] = sum_c W[o,c]*hT[p,c].  Q,K stored [HW,C]; V stored [C,HW].
    wmma_gemm_kernel<false, true, true, false>
        <<<dim3(HW / 64, C / 256, Bn), 256, 0, stream>>>(
            wq16, C, 0ull, hT, C, (ull)HW * C,
            qT, C, (ull)HW * C, bq, nullptr, 0ull, 1.0f, C);
    wmma_gemm_kernel<false, true, true, false>
        <<<dim3(HW / 64, C / 256, Bn), 256, 0, stream>>>(
            wk16, C, 0ull, hT, C, (ull)HW * C,
            kT, C, (ull)HW * C, bk, nullptr, 0ull, 1.0f, C);
    wmma_gemm_kernel<false, false, true, false>
        <<<dim3(HW / 64, C / 256, Bn), 256, 0, stream>>>(
            wv16, C, 0ull, hT, C, (ull)HW * C,
            vbuf, HW, (ull)C * HW, bv, nullptr, 0ull, 1.0f, C);

    // 4) S[i,j] = scale * sum_c qT[i,c]*kT[j,c]  -> f16 logits [B,HW,HW]
    const float scale = 0.044194173824159216f;  // 1/sqrt(512)
    wmma_gemm_kernel<false, false, false, false>
        <<<dim3(HW / 64, HW / 256, Bn), 256, 0, stream>>>(
            qT, C, (ull)HW * C, kT, C, (ull)HW * C,
            attn, HW, (ull)HW * HW, nullptr, nullptr, 0ull, scale, C);

    // 5) softmax rows in place
    softmax_rows_kernel<<<Bn * HW, 256, 0, stream>>>(attn, HW);

    // 6) ao[c,i] = sum_j v[c,j] * P[i,j]; stored transposed aoT[i,c]
    wmma_gemm_kernel<false, true, false, false>
        <<<dim3(HW / 64, C / 256, Bn), 256, 0, stream>>>(
            vbuf, HW, (ull)C * HW, attn, HW, (ull)HW * HW,
            aoT, C, (ull)HW * C, nullptr, nullptr, 0ull, 1.0f, HW);

    // 7) out[c,p] = x[c,p] + sum_o' wo[c,o']*aoT[p,o'] + bo[c]  -> f32 d_out
    wmma_gemm_kernel<true, false, true, true>
        <<<dim3(HW / 64, C / 256, Bn), 256, 0, stream>>>(
            wo16, C, 0ull, aoT, C, (ull)HW * C,
            d_out, HW, (ull)C * HW, bo, x, (ull)C * HW, 1.0f, C);
}
